// GATModel_31190052503871
// MI455X (gfx1250) — matrix operand order repaired and checked by reading, not compiled, not run on hardware
//
#include <hip/hip_runtime.h>
#include <hip/hip_bf16.h>

typedef __attribute__((ext_vector_type(16))) __bf16 v16bf;
typedef __attribute__((ext_vector_type(8)))  __bf16 v8bf;
typedef __attribute__((ext_vector_type(8)))  float  v8f;

#define NEG_SLOPE 0.2f

// ---------------------------------------------------------------------------
// fp32 -> bf16 elementwise convert (A operands)
// ---------------------------------------------------------------------------
__global__ void cvt_f32_bf16(const float* __restrict__ in, __bf16* __restrict__ out,
                             long n) {
  long i = (long)blockIdx.x * blockDim.x + threadIdx.x;
  if (i < n) out[i] = (__bf16)in[i];
}

// ---------------------------------------------------------------------------
// Pack B[K,N] (f32 row-major) into WMMA B fragments:
//   Bf[((s*(N/16)+t)*32 + lane)*16 + e] = bf16(B[(32s + (e<8 ? 8*half+e : 16+8*half+e-8))*N + 16t + lm])
// so each lane's 16-element fragment is one contiguous 32-byte run.
// One thread per (s, t, lane); weights are tiny so this is free.
// ---------------------------------------------------------------------------
__global__ void pack_B_frag(const float* __restrict__ B, __bf16* __restrict__ Bf,
                            int K, int N) {
  int idx = blockIdx.x * blockDim.x + threadIdx.x;
  int total = (K / 32) * (N / 16) * 32;
  if (idx >= total) return;
  int lane = idx & 31;
  int r    = idx >> 5;
  int t    = r % (N / 16);
  int s    = r / (N / 16);
  int half = lane >> 4, lm = lane & 15;
  __bf16* o = Bf + (size_t)idx * 16;
  #pragma unroll
  for (int e = 0; e < 16; ++e)
    o[e] = (__bf16)B[(size_t)(32 * s + (e < 8 ? 8 * half + e : 16 + 8 * half + (e - 8))) * N + 16 * t + lm];
}

// ---------------------------------------------------------------------------
// WMMA GEMM: C[M,N] = A[M,K] * B[K,N]; A row-major bf16, B pre-packed frags.
// One wave32 computes a 16x64 C tile: 4 accumulators share one A fragment per
// K-step; inner loop is pure b128 loads + 4 back-to-back WMMAs.
// A (16x32 bf16): lane holds row lane%16; elems 0..7 -> K=8*half+e,
//                 elems 8..15 -> K=16+8*half+(e-8)   (half = lane>>4)
// C/D (16x16 f32): vgpr v -> row v+8*half, col lane%16
// Requires M%16==0, N%64==0, K%32==0 (holds: M=50000, K=128/256, N=256/64).
// ---------------------------------------------------------------------------
template <int K>
__global__ void gemm_wmma_pk(const __bf16* __restrict__ A,
                             const __bf16* __restrict__ Bf,
                             float* __restrict__ C,
                             int M, int N) {
  const int lane = threadIdx.x & 31;
  const int wave = threadIdx.x >> 5;
  const int nG   = N >> 6;                       // 64-wide column groups
  const int tile = blockIdx.x * (blockDim.x >> 5) + wave;
  if (tile >= (M >> 4) * nG) return;
  const int mb   = (tile / nG) << 4;
  const int nb0  = (tile % nG) << 6;
  const int half = lane >> 4;
  const int lm   = lane & 15;
  const int nT16 = N >> 4;

  v8f acc[4] = {};
  const __bf16* Arow = A + (size_t)(mb + lm) * K;
  #pragma unroll
  for (int s = 0; s < K / 32; ++s) {
    const int kb = 32 * s;
    // A fragment: two contiguous 16B chunks
    v8bf alo = *(const v8bf*)(Arow + kb + 8 * half);
    v8bf ahi = *(const v8bf*)(Arow + kb + 16 + 8 * half);
    v16bf a  = __builtin_shufflevector(alo, ahi, 0, 1, 2, 3, 4, 5, 6, 7,
                                       8, 9, 10, 11, 12, 13, 14, 15);
    #pragma unroll
    for (int t = 0; t < 4; ++t) {
      const __bf16* bp = Bf + (((size_t)s * nT16 + (nb0 >> 4) + t) * 32 + lane) * 16;
      v8bf blo = *(const v8bf*)bp;
      v8bf bhi = *(const v8bf*)(bp + 8);
      v16bf b  = __builtin_shufflevector(blo, bhi, 0, 1, 2, 3, 4, 5, 6, 7,
                                         8, 9, 10, 11, 12, 13, 14, 15);
      acc[t] = __builtin_amdgcn_wmma_f32_16x16x32_bf16(
          /*neg_a=*/false, a, /*neg_b=*/false, b,
          /*c_mod=*/(short)0, acc[t], /*reuse_a=*/false, /*reuse_b=*/false);
    }
  }
  #pragma unroll
  for (int t = 0; t < 4; ++t)
    #pragma unroll
    for (int v = 0; v < 8; ++v)
      C[(size_t)(mb + v + 8 * half) * N + (nb0 + 16 * t + lm)] = acc[t][v];
}

// ---------------------------------------------------------------------------
// Helpers / edge-phase kernels (h tables fit in 192MB L2 -> random access OK)
// ---------------------------------------------------------------------------
__global__ void fill_f32(float* __restrict__ p, float v, long n) {
  long i = (long)blockIdx.x * blockDim.x + threadIdx.x;
  if (i < n) p[i] = v;
}

__global__ void att_scores(const float* __restrict__ h,
                           const float* __restrict__ att_s,
                           const float* __restrict__ att_d,
                           float* __restrict__ as_, float* __restrict__ ad_,
                           int N, int H, int Cc) {
  int t = blockIdx.x * blockDim.x + threadIdx.x;
  if (t >= N * H) return;
  int n = t / H, hh = t % H;
  const float* hp  = h + ((size_t)n * H + hh) * Cc;
  const float* asp = att_s + (size_t)hh * Cc;
  const float* adp = att_d + (size_t)hh * Cc;
  float s = 0.f, d = 0.f;
  for (int c = 0; c < Cc; ++c) { s += hp[c] * asp[c]; d += hp[c] * adp[c]; }
  as_[t] = s; ad_[t] = d;
}

__device__ __forceinline__ void edge_sd(int i, int E, const int* __restrict__ ei,
                                        int& s, int& d) {
  if (i < E) { s = ei[i]; d = ei[E + i]; }
  else       { s = i - E; d = i - E; }      // self-loops appended
}

__device__ __forceinline__ void atomicMaxF(float* addr, float val) {
  if (val >= 0.f) atomicMax((int*)addr, __float_as_int(val));
  else            atomicMin((unsigned int*)addr, __float_as_uint(val));
}

__global__ void edge_pass1(const int* __restrict__ ei, int E, int Etot, int H,
                           const float* __restrict__ as_, const float* __restrict__ ad_,
                           float* __restrict__ e, float* __restrict__ m) {
  int i = blockIdx.x * blockDim.x + threadIdx.x;
  if (i >= Etot) return;
  int s, d; edge_sd(i, E, ei, s, d);
  for (int h = 0; h < H; ++h) {
    float v = as_[s * H + h] + ad_[d * H + h];
    v = v > 0.f ? v : NEG_SLOPE * v;
    e[(size_t)i * H + h] = v;
    atomicMaxF(&m[d * H + h], v);
  }
}

__global__ void edge_pass2(const int* __restrict__ ei, int E, int Etot, int H,
                           const float* __restrict__ m,
                           float* __restrict__ e, float* __restrict__ dn) {
  int i = blockIdx.x * blockDim.x + threadIdx.x;
  if (i >= Etot) return;
  int s, d; edge_sd(i, E, ei, s, d);
  for (int h = 0; h < H; ++h) {
    float ex = __expf(e[(size_t)i * H + h] - m[d * H + h]);
    e[(size_t)i * H + h] = ex;
    atomicAdd(&dn[d * H + h], ex);
  }
}

__global__ void edge_alpha(const int* __restrict__ ei, int E, int Etot, int H,
                           const float* __restrict__ dn, float* __restrict__ e) {
  int i = blockIdx.x * blockDim.x + threadIdx.x;
  if (i >= Etot) return;
  int s, d; edge_sd(i, E, ei, s, d);
  for (int h = 0; h < H; ++h)
    e[(size_t)i * H + h] /= dn[d * H + h];
}

__global__ void edge_aggregate(const int* __restrict__ ei, int E, int Etot,
                               int H, int Cc,
                               const float* __restrict__ alpha,
                               const float* __restrict__ h,
                               float* __restrict__ out) {
  long t = (long)blockIdx.x * blockDim.x + threadIdx.x;
  if (t >= (long)Etot * Cc) return;
  int i = (int)(t / Cc), c = (int)(t % Cc);
  int s, d; edge_sd(i, E, ei, s, d);
  for (int hh = 0; hh < H; ++hh) {
    float a = alpha[(size_t)i * H + hh];
    atomicAdd(&out[((size_t)d * H + hh) * Cc + c],
              a * h[((size_t)s * H + hh) * Cc + c]);
  }
}

__global__ void bias_elu(float* __restrict__ x, const float* __restrict__ b,
                         int Cc, long n) {
  long i = (long)blockIdx.x * blockDim.x + threadIdx.x;
  if (i >= n) return;
  float v = x[i] + b[i % Cc];
  x[i] = v > 0.f ? v : (__expf(v) - 1.f);
}

__global__ void bias_add(float* __restrict__ x, const float* __restrict__ b,
                         int Cc, long n) {
  long i = (long)blockIdx.x * blockDim.x + threadIdx.x;
  if (i >= n) return;
  x[i] += b[i % Cc];
}

// ---------------------------------------------------------------------------
extern "C" void kernel_launch(void* const* d_in, const int* in_sizes, int n_in,
                              void* d_out, int out_size, void* d_ws, size_t ws_size,
                              hipStream_t stream) {
  const float* x    = (const float*)d_in[0];
  const int*   ei   = (const int*)  d_in[1];
  const float* W1   = (const float*)d_in[2];
  const float* aS1  = (const float*)d_in[3];
  const float* aD1  = (const float*)d_in[4];
  const float* b1   = (const float*)d_in[5];
  const float* W2   = (const float*)d_in[6];
  const float* aS2  = (const float*)d_in[7];
  const float* aD2  = (const float*)d_in[8];
  const float* b2   = (const float*)d_in[9];
  float* out = (float*)d_out;

  const int N    = in_sizes[0] / 128;   // 50000
  const int E    = in_sizes[1] / 2;     // 800000
  const int Etot = E + N;
  const int IN = 128, H1 = 4, C1 = 64, F1 = H1 * C1 /*256*/, C2 = 64;

  // workspace layout (float units; bf16 arrays count half)
  float* ws   = (float*)d_ws;
  size_t o    = 0;
  float* h1   = ws + o; o += (size_t)N * F1;     // x@W1            [N,256]
  float* agg1 = ws + o; o += (size_t)N * F1;     // layer1 out / h2in (ELU in place)
  float* h2   = ws + o; o += (size_t)N * C2;     // h2in@W2         [N,64]
  float* as1  = ws + o; o += (size_t)N * H1;
  float* ad1  = ws + o; o += (size_t)N * H1;
  float* m1   = ws + o; o += (size_t)N * H1;
  float* dn1  = ws + o; o += (size_t)N * H1;
  float* e1   = ws + o; o += (size_t)Etot * H1;
  float* as2  = ws + o; o += (size_t)N;
  float* ad2  = ws + o; o += (size_t)N;
  float* m2   = ws + o; o += (size_t)N;
  float* dn2  = ws + o; o += (size_t)N;
  float* e2   = ws + o; o += (size_t)Etot;
  __bf16* xbf  = (__bf16*)(ws + o); o += (size_t)N * IN / 2;   // x as bf16 [N,128]
  __bf16* h2bf = (__bf16*)(ws + o); o += (size_t)N * F1 / 2;   // h2in bf16 [N,256]
  __bf16* Bf1  = (__bf16*)(ws + o); o += (size_t)(IN / 32) * (F1 / 16) * 512 / 2;
  __bf16* Bf2  = (__bf16*)(ws + o); o += (size_t)(F1 / 32) * (C2 / 16) * 512 / 2;

  const int TB = 256;
  auto blk1 = [&](long n) { return (unsigned)((n + TB - 1) / TB); };
  const float NINF = -__builtin_inff();

  // -------- layer 1 --------
  cvt_f32_bf16<<<blk1((long)N * IN), TB, 0, stream>>>(x, xbf, (long)N * IN);
  pack_B_frag<<<blk1((IN / 32) * (F1 / 16) * 32), TB, 0, stream>>>(W1, Bf1, IN, F1);
  {
    int tiles = (N / 16) * (F1 / 64);            // 16x64 tile per wave
    gemm_wmma_pk<128><<<(tiles + 3) / 4, 128, 0, stream>>>(xbf, Bf1, h1, N, F1);
  }
  att_scores<<<blk1((long)N * H1), TB, 0, stream>>>(h1, aS1, aD1, as1, ad1, N, H1, C1);
  fill_f32<<<blk1((long)N * H1), TB, 0, stream>>>(m1,  NINF, (long)N * H1);
  fill_f32<<<blk1((long)N * H1), TB, 0, stream>>>(dn1, 0.f,  (long)N * H1);
  fill_f32<<<blk1((long)N * F1), TB, 0, stream>>>(agg1, 0.f, (long)N * F1);
  edge_pass1<<<blk1(Etot), TB, 0, stream>>>(ei, E, Etot, H1, as1, ad1, e1, m1);
  edge_pass2<<<blk1(Etot), TB, 0, stream>>>(ei, E, Etot, H1, m1, e1, dn1);
  edge_alpha<<<blk1(Etot), TB, 0, stream>>>(ei, E, Etot, H1, dn1, e1);
  edge_aggregate<<<blk1((long)Etot * C1), TB, 0, stream>>>(ei, E, Etot, H1, C1,
                                                           e1, h1, agg1);
  bias_elu<<<blk1((long)N * F1), TB, 0, stream>>>(agg1, b1, F1, (long)N * F1);

  // -------- layer 2 --------
  cvt_f32_bf16<<<blk1((long)N * F1), TB, 0, stream>>>(agg1, h2bf, (long)N * F1);
  pack_B_frag<<<blk1((F1 / 32) * (C2 / 16) * 32), TB, 0, stream>>>(W2, Bf2, F1, C2);
  {
    int tiles = (N / 16) * (C2 / 64);
    gemm_wmma_pk<256><<<(tiles + 3) / 4, 128, 0, stream>>>(h2bf, Bf2, h2, N, C2);
  }
  att_scores<<<blk1((long)N), TB, 0, stream>>>(h2, aS2, aD2, as2, ad2, N, 1, C2);
  fill_f32<<<blk1((long)N), TB, 0, stream>>>(m2,  NINF, (long)N);
  fill_f32<<<blk1((long)N), TB, 0, stream>>>(dn2, 0.f,  (long)N);
  fill_f32<<<blk1((long)N * C2), TB, 0, stream>>>(out, 0.f, (long)N * C2);
  edge_pass1<<<blk1(Etot), TB, 0, stream>>>(ei, E, Etot, 1, as2, ad2, e2, m2);
  edge_pass2<<<blk1(Etot), TB, 0, stream>>>(ei, E, Etot, 1, m2, e2, dn2);
  edge_alpha<<<blk1(Etot), TB, 0, stream>>>(ei, E, Etot, 1, dn2, e2);
  edge_aggregate<<<blk1((long)Etot * C2), TB, 0, stream>>>(ei, E, Etot, 1, C2,
                                                           e2, h2, out);
  bias_add<<<blk1((long)N * C2), TB, 0, stream>>>(out, b2, C2, (long)N * C2);
}
